// Transformer_Block_32461362823502
// MI455X (gfx1250) — matrix operand order repaired and checked by reading, not compiled, not run on hardware
//
#include <hip/hip_runtime.h>

// ---------------------------------------------------------------------------
// Types for CDNA5 WMMA
// ---------------------------------------------------------------------------
typedef __attribute__((ext_vector_type(16))) __bf16 v16bf;
typedef __attribute__((ext_vector_type(8)))  float  v8f;
typedef int v4i_ __attribute__((vector_size(16)));   // matches async-LDS builtin

#define TOK   4096          // tokens per batch (N)
#define CH    128           // channels (C)
#define ROWS  8192          // B * TOK
#define HDIM  32            // head dim
#define NHEAD 4
#define C2    256           // C * EXP

// Async global->LDS path (gfx1250 GLOBAL_LOAD_ASYNC_TO_LDS_*, ASYNCcnt)
#if __has_builtin(__builtin_amdgcn_global_load_async_to_lds_b128) && \
    __has_builtin(__builtin_amdgcn_s_wait_asynccnt)
#define ASYNC_LDS 1
#else
#define ASYNC_LDS 0
#endif

__device__ __forceinline__ __bf16 f2bf(float f) {
  unsigned u = __float_as_uint(f);
  unsigned r = u + 0x7FFFu + ((u >> 16) & 1u);   // round-to-nearest-even
  unsigned short h = (unsigned short)(r >> 16);
  return __builtin_bit_cast(__bf16, h);
}

#if ASYNC_LDS
// copy 16 bytes (8 bf16) global -> LDS via the async data path
__device__ __forceinline__ void g2lds16(const __bf16* g, __bf16* l) {
  __builtin_amdgcn_global_load_async_to_lds_b128(
      (__attribute__((address_space(1))) v4i_*)g,
      (__attribute__((address_space(3))) v4i_*)l, 0, 0);
}
#endif

// ---------------------------------------------------------------------------
// f32 -> bf16 conversion (weights)
// ---------------------------------------------------------------------------
__global__ void cvt_bf16_kernel(const float* __restrict__ in,
                                __bf16* __restrict__ out, int n) {
  int i = blockIdx.x * blockDim.x + threadIdx.x;
  if (i < n) out[i] = f2bf(in[i]);
}

// ---------------------------------------------------------------------------
// LayerNorm over C=128, one row per 128-thread block, bf16 output
// ---------------------------------------------------------------------------
__global__ __launch_bounds__(128)
void ln_bf16_kernel(const float* __restrict__ x, const float* __restrict__ w,
                    const float* __restrict__ b, __bf16* __restrict__ out) {
  const int row  = blockIdx.x;
  const int c    = threadIdx.x;
  const int lane = threadIdx.x & 31;
  const int wv   = threadIdx.x >> 5;
  __shared__ float red1[4], red2[4];

  float v = x[(size_t)row * CH + c];
  float s = v;
#pragma unroll
  for (int m = 16; m > 0; m >>= 1) s += __shfl_xor(s, m, 32);
  if (lane == 0) red1[wv] = s;
  __syncthreads();
  float mu = (red1[0] + red1[1] + red1[2] + red1[3]) * (1.0f / CH);
  float d  = v - mu;
  float s2 = d * d;
#pragma unroll
  for (int m = 16; m > 0; m >>= 1) s2 += __shfl_xor(s2, m, 32);
  if (lane == 0) red2[wv] = s2;
  __syncthreads();
  float var = (red2[0] + red2[1] + red2[2] + red2[3]) * (1.0f / CH);
  float y = d * rsqrtf(var + 1e-5f) * w[c] + b[c];
  out[(size_t)row * CH + c] = f2bf(y);
}

// ---------------------------------------------------------------------------
// Tiled bf16 WMMA GEMM:  out[M,N] = A[M,K] * W[N,K]^T (+bias) (+residual)
// Block = 128 threads (4 waves), 64x64 output tile, K stepped by 32.
// Double-buffered LDS: next tile's async loads overlap current tile's WMMAs.
// ---------------------------------------------------------------------------
__global__ __launch_bounds__(128)
void gemm_bf16_wmma(const __bf16* __restrict__ A, const __bf16* __restrict__ W,
                    const float* __restrict__ bias, const float* __restrict__ res,
                    float* __restrict__ outF, __bf16* __restrict__ outB,
                    int M, int N, int K) {
  __shared__ __bf16 As[2][64][32];
  __shared__ __bf16 Ws[2][64][32];

  const int tid  = threadIdx.x;
  const int wv   = tid >> 5;
  const int lane = tid & 31;
  const int row0 = blockIdx.x * 64;
  const int col0 = blockIdx.y * 64;

  const v8f vzero = {0.f, 0.f, 0.f, 0.f, 0.f, 0.f, 0.f, 0.f};
  v8f acc[4] = {vzero, vzero, vzero, vzero};

  const int lr = tid >> 1;        // tile row this thread stages (0..63)
  const int lk = (tid & 1) * 16;  // k sub-chunk (0 or 16)
  const __bf16* gA = &A[(size_t)(row0 + lr) * K + lk];
  const __bf16* gW = &W[(size_t)(col0 + lr) * K + lk];
  const int nsteps = K >> 5;

#if ASYNC_LDS
  // prologue: tile 0 in flight
  g2lds16(gA,     &As[0][lr][lk]);
  g2lds16(gA + 8, &As[0][lr][lk + 8]);
  g2lds16(gW,     &Ws[0][lr][lk]);
  g2lds16(gW + 8, &Ws[0][lr][lk + 8]);
#endif

  for (int s = 0; s < nsteps; ++s) {
    const int buf = s & 1;
    const int nb  = buf ^ 1;
    const int k0  = s * 32;
#if ASYNC_LDS
    if (s + 1 < nsteps) {
      // issue next tile while current is consumed below
      g2lds16(gA + k0 + 32, &As[nb][lr][lk]);
      g2lds16(gA + k0 + 40, &As[nb][lr][lk + 8]);
      g2lds16(gW + k0 + 32, &Ws[nb][lr][lk]);
      g2lds16(gW + k0 + 40, &Ws[nb][lr][lk + 8]);
      if (s + 2 < nsteps) {
        __builtin_prefetch(gA + k0 + 64, 0, 3);
        __builtin_prefetch(gW + k0 + 64, 0, 3);
      }
      __builtin_amdgcn_s_wait_asynccnt(4);   // allow the 4 just-issued in flight
    } else {
      __builtin_amdgcn_s_wait_asynccnt(0);
    }
    __syncthreads();
#else
    float4 av0 = reinterpret_cast<const float4*>(gA + k0)[0];
    float4 av1 = reinterpret_cast<const float4*>(gA + k0)[1];
    float4 wv0 = reinterpret_cast<const float4*>(gW + k0)[0];
    float4 wv1 = reinterpret_cast<const float4*>(gW + k0)[1];
    __syncthreads();
    *reinterpret_cast<float4*>(&As[buf][lr][lk])     = av0;
    *reinterpret_cast<float4*>(&As[buf][lr][lk + 8]) = av1;
    *reinterpret_cast<float4*>(&Ws[buf][lr][lk])     = wv0;
    *reinterpret_cast<float4*>(&Ws[buf][lr][lk + 8]) = wv1;
    __syncthreads();
#endif

    // A fragment (16x32): lanes 0-15 K=0..15, lanes 16-31 K=16..31
    v16bf afrag;
    const int ar  = wv * 16 + (lane & 15);
    const int akb = (lane >> 4) * 16;
#pragma unroll
    for (int i = 0; i < 8; ++i) afrag[i]     = As[buf][ar][akb + i];
#pragma unroll
    for (int i = 0; i < 8; ++i) afrag[8 + i] = As[buf][ar][akb + 8 + i];

#pragma unroll
    for (int ct = 0; ct < 4; ++ct) {
      // B fragment (32x16): B[k][n] = W[n][k]; lanes 0-15 K=0..15, 16-31 K=16..31
      v16bf bfrag;
      const int bc  = ct * 16 + (lane & 15);
      const int bkb = (lane >> 4) * 16;
#pragma unroll
      for (int i = 0; i < 16; ++i) bfrag[i] = Ws[buf][bc][bkb + i];
      acc[ct] = __builtin_amdgcn_wmma_f32_16x16x32_bf16(
          false, afrag, false, bfrag, (short)0, acc[ct], false, false);
    }
#if ASYNC_LDS
    __syncthreads();   // readers done before next iteration overwrites buf^1
#endif
  }

  // epilogue: C/D layout row = (lane>>4)*8 + i, col = lane&15
  const int orow = row0 + wv * 16 + (lane >> 4) * 8;
#pragma unroll
  for (int ct = 0; ct < 4; ++ct) {
    const int oc = col0 + ct * 16 + (lane & 15);
    const float bvv = bias ? bias[oc] : 0.0f;
#pragma unroll
    for (int i = 0; i < 8; ++i) {
      const int r = orow + i;
      float v = acc[ct][i] + bvv;
      if (res)  v += res[(size_t)r * N + oc];
      if (outF) outF[(size_t)r * N + oc] = v;
      if (outB) outB[(size_t)r * N + oc] = f2bf(v);
    }
  }
}

// ---------------------------------------------------------------------------
// Flash attention, bf16 WMMA, f32 online softmax, double-buffered K/V tiles.
// grid.x = B*NHEAD, grid.y = TOK/64. Block = 128 threads (4 waves).
// K staged [key][d] (async); V staged TRANSPOSED [d][key] so the P*V
// B-fragment is a contiguous 16-element run -> 2x ds_load_b128.
// Pt is wave-private: per-wave in-order LDS makes a block barrier unnecessary.
// ---------------------------------------------------------------------------
__global__ __launch_bounds__(128)
void attn_wmma_kernel(const __bf16* __restrict__ Q, const __bf16* __restrict__ KV,
                      __bf16* __restrict__ O, float scale) {
  const int bh = blockIdx.x;
  const int b  = bh >> 2;
  const int h  = bh & 3;
  const int q0 = blockIdx.y * 64;

  __shared__ __bf16 Kt[2][64][32];
  __shared__ __bf16 VtT[2][32][72];     // [d][key], padded: rows 144B (16B align)
  __shared__ __bf16 Pt[4][16][64];

  const int tid  = threadIdx.x;
  const int wv   = tid >> 5;
  const int lane = tid & 31;

  // Q A-fragment for this wave's 16 rows (K = HDIM = 32 -> single fragment)
  const int qrow = q0 + wv * 16 + (lane & 15);
  const int qkb  = (lane >> 4) * 16;
  const __bf16* qbase = &Q[((size_t)b * TOK + qrow) * CH + h * HDIM];
  v16bf qfrag;
#pragma unroll
  for (int i = 0; i < 8; ++i) qfrag[i]     = qbase[qkb + i];
#pragma unroll
  for (int i = 0; i < 8; ++i) qfrag[8 + i] = qbase[qkb + 8 + i];

  const v8f vzero = {0.f, 0.f, 0.f, 0.f, 0.f, 0.f, 0.f, 0.f};
  v8f o0 = vzero, o1 = vzero;
  float mrow[8], lrow[8];
#pragma unroll
  for (int i = 0; i < 8; ++i) { mrow[i] = -3.0e38f; lrow[i] = 0.0f; }

  const int lr = tid >> 1;         // key row staged by this thread (0..63)
  const int lk = (tid & 1) * 16;   // d sub-chunk (0 or 16)
  const __bf16* kvbase = &KV[((size_t)b * TOK + lr) * C2 + h * HDIM + lk];
  const size_t tstride = (size_t)64 * C2;    // elements between key tiles

#if ASYNC_LDS
  // prologue: K tile 0 in flight
  g2lds16(kvbase,     &Kt[0][lr][lk]);
  g2lds16(kvbase + 8, &Kt[0][lr][lk + 8]);
#else
  float4 kc0 = *reinterpret_cast<const float4*>(kvbase);
  float4 kc1 = *reinterpret_cast<const float4*>(kvbase + 8);
#endif
  // V tile 0 -> registers
  float4 vc0 = *reinterpret_cast<const float4*>(kvbase + CH);
  float4 vc1 = *reinterpret_cast<const float4*>(kvbase + CH + 8);

  for (int j0 = 0; j0 < TOK; j0 += 64) {
    const int  buf     = (j0 >> 6) & 1;
    const int  nb      = buf ^ 1;
    const bool hasnext = (j0 + 64 < TOK);
    const __bf16* kvnext = kvbase + (size_t)((j0 >> 6) + 1) * tstride;

    float4 vn0 = vc0, vn1 = vc1;
#if !ASYNC_LDS
    float4 kn0 = kc0, kn1 = kc1;
#endif
    if (hasnext) {
#if ASYNC_LDS
      g2lds16(kvnext,     &Kt[nb][lr][lk]);      // overlap with compute below
      g2lds16(kvnext + 8, &Kt[nb][lr][lk + 8]);
#else
      kn0 = *reinterpret_cast<const float4*>(kvnext);
      kn1 = *reinterpret_cast<const float4*>(kvnext + 8);
#endif
      vn0 = *reinterpret_cast<const float4*>(kvnext + CH);
      vn1 = *reinterpret_cast<const float4*>(kvnext + CH + 8);
    }

    // stage current tile (safe: buf last read 2 iterations ago, barriers since)
#if !ASYNC_LDS
    *reinterpret_cast<float4*>(&Kt[buf][lr][lk])     = kc0;
    *reinterpret_cast<float4*>(&Kt[buf][lr][lk + 8]) = kc1;
#endif
    {
      __bf16 vtmp[16];
      *reinterpret_cast<float4*>(&vtmp[0]) = vc0;
      *reinterpret_cast<float4*>(&vtmp[8]) = vc1;
#pragma unroll
      for (int i = 0; i < 16; ++i) VtT[buf][lk + i][lr] = vtmp[i];   // transpose
    }
#if ASYNC_LDS
    if (hasnext) __builtin_amdgcn_s_wait_asynccnt(2);  // drain current, keep next in flight
    else         __builtin_amdgcn_s_wait_asynccnt(0);
#endif
    __syncthreads();   // barrier A: current tiles ready

    // S = Q * K^T   (16 x 64 per wave, 4 WMMAs)
    float Sf[4][8];
#pragma unroll
    for (int ct = 0; ct < 4; ++ct) {
      v16bf kfrag;
      const int kc  = ct * 16 + (lane & 15);
      const int kkb = (lane >> 4) * 16;
#pragma unroll
      for (int i = 0; i < 16; ++i) kfrag[i] = Kt[buf][kc][kkb + i];
      v8f s = __builtin_amdgcn_wmma_f32_16x16x32_bf16(
          false, qfrag, false, kfrag, (short)0, vzero, false, false);
#pragma unroll
      for (int i = 0; i < 8; ++i) Sf[ct][i] = s[i] * scale;
    }

    // online softmax (rows live across 16 lanes of a half-wave)
    float tmax[8];
#pragma unroll
    for (int i = 0; i < 8; ++i) tmax[i] = -3.0e38f;
#pragma unroll
    for (int ct = 0; ct < 4; ++ct)
#pragma unroll
      for (int i = 0; i < 8; ++i) tmax[i] = fmaxf(tmax[i], Sf[ct][i]);
#pragma unroll
    for (int i = 0; i < 8; ++i)
#pragma unroll
      for (int m = 8; m > 0; m >>= 1) tmax[i] = fmaxf(tmax[i], __shfl_xor(tmax[i], m, 16));

    float mnew[8], alpha[8], psum[8];
#pragma unroll
    for (int i = 0; i < 8; ++i) {
      mnew[i]  = fmaxf(mrow[i], tmax[i]);
      alpha[i] = __expf(mrow[i] - mnew[i]);
      mrow[i]  = mnew[i];
      psum[i]  = 0.0f;
    }

    const int prb = (lane >> 4) * 8;
    const int pcb = lane & 15;
#pragma unroll
    for (int ct = 0; ct < 4; ++ct)
#pragma unroll
      for (int i = 0; i < 8; ++i) {
        float p = __expf(Sf[ct][i] - mnew[i]);
        psum[i] += p;
        Pt[wv][prb + i][ct * 16 + pcb] = f2bf(p);
      }
#pragma unroll
    for (int i = 0; i < 8; ++i)
#pragma unroll
      for (int m = 8; m > 0; m >>= 1) psum[i] += __shfl_xor(psum[i], m, 16);
#pragma unroll
    for (int i = 0; i < 8; ++i) {
      lrow[i] = lrow[i] * alpha[i] + psum[i];
      o0[i] *= alpha[i];
      o1[i] *= alpha[i];
    }
    // no block barrier: Pt strip is wave-private, LDS ops in-order per wave

    // O += P * V   (16x64 @ 64x32 -> 2 K-steps x 2 col tiles)
#pragma unroll
    for (int ks = 0; ks < 2; ++ks) {
      v16bf pfrag;
      const int pr  = lane & 15;
      const int pkb = (lane >> 4) * 16;
#pragma unroll
      for (int i = 0; i < 8; ++i) pfrag[i]     = Pt[wv][pr][ks * 32 + pkb + i];
#pragma unroll
      for (int i = 0; i < 8; ++i) pfrag[8 + i] = Pt[wv][pr][ks * 32 + pkb + 8 + i];
#pragma unroll
      for (int cv = 0; cv < 2; ++cv) {
        // B[k][d]: contiguous in VtT[d][key] -> 2x ds_load_b128
        v16bf vfrag;
        const int vc  = cv * 16 + (lane & 15);
        const int vkb = (lane >> 4) * 16;
#pragma unroll
        for (int i = 0; i < 16; ++i) vfrag[i] = VtT[buf][vc][ks * 32 + vkb + i];
        if (cv == 0)
          o0 = __builtin_amdgcn_wmma_f32_16x16x32_bf16(
              false, pfrag, false, vfrag, (short)0, o0, false, false);
        else
          o1 = __builtin_amdgcn_wmma_f32_16x16x32_bf16(
              false, pfrag, false, vfrag, (short)0, o1, false, false);
      }
    }
    __syncthreads();   // barrier B: reads done before next iteration's staging

    vc0 = vn0; vc1 = vn1;
#if !ASYNC_LDS
    kc0 = kn0; kc1 = kn1;
#endif
  }

  // normalize + store
  const int orow = q0 + wv * 16 + (lane >> 4) * 8;
  const int dlo  = lane & 15;
#pragma unroll
  for (int i = 0; i < 8; ++i) {
    const float inv = 1.0f / lrow[i];
    const size_t base = ((size_t)b * TOK + orow + i) * CH + h * HDIM;
    O[base + dlo]      = f2bf(o0[i] * inv);
    O[base + 16 + dlo] = f2bf(o1[i] * inv);
  }
}

// ---------------------------------------------------------------------------
// Depthwise 3x3 + 5x5 + identity sum, NHWC [ROWS, 256] f32
// ---------------------------------------------------------------------------
__global__ __launch_bounds__(256)
void dwconv_kernel(const float* __restrict__ x1,
                   const float* __restrict__ w3, const float* __restrict__ b3,
                   const float* __restrict__ w5, const float* __restrict__ b5,
                   float* __restrict__ s) {
  const size_t idx = (size_t)blockIdx.x * blockDim.x + threadIdx.x;
  if (idx >= (size_t)2 * 64 * 64 * C2) return;
  const int c  = idx & 255;
  const int wq = (idx >> 8) & 63;
  const int hq = (idx >> 14) & 63;
  const int bb = (int)(idx >> 20);
  const float* base = x1 + (size_t)bb * 4096 * C2;

  float acc3 = b3[c], acc5 = b5[c];
#pragma unroll
  for (int dy = -2; dy <= 2; ++dy) {
    const int yy = hq + dy;
    if (yy < 0 || yy >= 64) continue;
#pragma unroll
    for (int dx = -2; dx <= 2; ++dx) {
      const int xx = wq + dx;
      if (xx < 0 || xx >= 64) continue;
      const float v = base[((size_t)yy * 64 + xx) * C2 + c];
      acc5 += v * w5[c * 25 + (dy + 2) * 5 + (dx + 2)];
      if (dy >= -1 && dy <= 1 && dx >= -1 && dx <= 1)
        acc3 += v * w3[c * 9 + (dy + 1) * 3 + (dx + 1)];
    }
  }
  const float center = base[((size_t)hq * 64 + wq) * C2 + c];
  s[idx] = center + acc3 + acc5;
}

// ---------------------------------------------------------------------------
// SimpleGate: gated[row, c] = s[row, c] * s[row, c+128]  -> bf16
// ---------------------------------------------------------------------------
__global__ __launch_bounds__(256)
void gate_kernel(const float* __restrict__ s, __bf16* __restrict__ g) {
  const size_t idx = (size_t)blockIdx.x * blockDim.x + threadIdx.x;
  if (idx >= (size_t)ROWS * CH) return;
  const int c = idx & 127;
  const size_t row = idx >> 7;
  g[idx] = f2bf(s[row * C2 + c] * s[row * C2 + CH + c]);
}

// ---------------------------------------------------------------------------
// Host orchestration
// ---------------------------------------------------------------------------
extern "C" void kernel_launch(void* const* d_in, const int* in_sizes, int n_in,
                              void* d_out, int out_size, void* d_ws, size_t ws_size,
                              hipStream_t stream) {
  (void)in_sizes; (void)n_in; (void)out_size; (void)ws_size;
  const float* x       = (const float*)d_in[0];
  const float* ln1_w   = (const float*)d_in[3];
  const float* ln1_b   = (const float*)d_in[4];
  const float* q_w     = (const float*)d_in[5];
  const float* kv_w    = (const float*)d_in[6];
  const float* proj_w  = (const float*)d_in[7];
  const float* proj_b  = (const float*)d_in[8];
  const float* ln2_w   = (const float*)d_in[9];
  const float* ln2_b   = (const float*)d_in[10];
  const float* conv1_w = (const float*)d_in[11];
  const float* conv1_b = (const float*)d_in[12];
  const float* conv33_w= (const float*)d_in[13];
  const float* conv33_b= (const float*)d_in[14];
  const float* conv55_w= (const float*)d_in[15];
  const float* conv55_b= (const float*)d_in[16];
  const float* conv4_w = (const float*)d_in[17];
  const float* conv4_b = (const float*)d_in[18];

  char* p = (char*)d_ws;
  auto alloc = [&](size_t bytes) -> void* {
    void* r = (void*)p;
    p += (bytes + 255) & ~(size_t)255;
    return r;
  };
  __bf16* wq_bf    = (__bf16*)alloc((size_t)CH * CH * 2);
  __bf16* wkv_bf   = (__bf16*)alloc((size_t)2 * CH * CH * 2);
  __bf16* wproj_bf = (__bf16*)alloc((size_t)CH * CH * 2);
  __bf16* wc1_bf   = (__bf16*)alloc((size_t)C2 * CH * 2);
  __bf16* wc4_bf   = (__bf16*)alloc((size_t)CH * CH * 2);
  __bf16* h_bf     = (__bf16*)alloc((size_t)ROWS * CH * 2);
  __bf16* q_bf     = (__bf16*)alloc((size_t)ROWS * CH * 2);
  __bf16* kv_bf    = (__bf16*)alloc((size_t)ROWS * C2 * 2);
  __bf16* attn_bf  = (__bf16*)alloc((size_t)ROWS * CH * 2);
  float*  xres     = (float*) alloc((size_t)ROWS * CH * 4);
  float*  x1f      = (float*) alloc((size_t)ROWS * C2 * 4);
  float*  sf       = (float*) alloc((size_t)ROWS * C2 * 4);
  __bf16* gated_bf = (__bf16*)alloc((size_t)ROWS * CH * 2);

  // weight conversions
  cvt_bf16_kernel<<<(CH*CH + 255) / 256, 256, 0, stream>>>(q_w, wq_bf, CH * CH);
  cvt_bf16_kernel<<<(2*CH*CH + 255) / 256, 256, 0, stream>>>(kv_w, wkv_bf, 2 * CH * CH);
  cvt_bf16_kernel<<<(CH*CH + 255) / 256, 256, 0, stream>>>(proj_w, wproj_bf, CH * CH);
  cvt_bf16_kernel<<<(C2*CH + 255) / 256, 256, 0, stream>>>(conv1_w, wc1_bf, C2 * CH);
  cvt_bf16_kernel<<<(CH*CH + 255) / 256, 256, 0, stream>>>(conv4_w, wc4_bf, CH * CH);

  // LN1
  ln_bf16_kernel<<<ROWS, 128, 0, stream>>>(x, ln1_w, ln1_b, h_bf);

  // q / kv projections
  gemm_bf16_wmma<<<dim3(ROWS / 64, CH / 64), 128, 0, stream>>>(
      h_bf, wq_bf, nullptr, nullptr, nullptr, q_bf, ROWS, CH, CH);
  gemm_bf16_wmma<<<dim3(ROWS / 64, C2 / 64), 128, 0, stream>>>(
      h_bf, wkv_bf, nullptr, nullptr, nullptr, kv_bf, ROWS, C2, CH);

  // flash attention
  const float scale = 0.17677669529663687f;  // 1/sqrt(32)
  attn_wmma_kernel<<<dim3(2 * NHEAD, TOK / 64), 128, 0, stream>>>(
      q_bf, kv_bf, attn_bf, scale);

  // proj + bias + residual(x) -> xres (f32)
  gemm_bf16_wmma<<<dim3(ROWS / 64, CH / 64), 128, 0, stream>>>(
      attn_bf, wproj_bf, proj_b, x, xres, nullptr, ROWS, CH, CH);

  // LN2
  ln_bf16_kernel<<<ROWS, 128, 0, stream>>>(xres, ln2_w, ln2_b, h_bf);

  // conv1 (1x1): [ROWS,128] -> [ROWS,256] f32 NHWC
  gemm_bf16_wmma<<<dim3(ROWS / 64, C2 / 64), 128, 0, stream>>>(
      h_bf, wc1_bf, conv1_b, nullptr, x1f, nullptr, ROWS, C2, CH);

  // dw3 + dw5 + identity
  dwconv_kernel<<<(2 * 64 * 64 * C2 + 255) / 256, 256, 0, stream>>>(
      x1f, conv33_w, conv33_b, conv55_w, conv55_b, sf);

  // SimpleGate
  gate_kernel<<<((size_t)ROWS * CH + 255) / 256, 256, 0, stream>>>(sf, gated_bf);

  // conv4 (1x1) + bias + residual(xres) -> d_out (f32)
  gemm_bf16_wmma<<<dim3(ROWS / 64, CH / 64), 128, 0, stream>>>(
      gated_bf, wc4_bf, conv4_b, xres, (float*)d_out, nullptr, ROWS, CH, CH);
}
